// GroupedExpert_3530463117617
// MI455X (gfx1250) — matrix operand order repaired and checked by reading, not compiled, not run on hardware
//
#include <hip/hip_runtime.h>

// MoE grouped FFN: out = (silu(x@w1[e]) * (x@w3[e])) @ w2[e], bf16 compute, fp32 out.
// E=8, D=4096, I=1024, T=32768 (T derived from in_sizes; D/I fixed per reference).
//
// MI455X strategy (compute-bound: ~0.83 TFLOP WMMA vs ~60us HBM floor):
//   Pre-pass: x -> bf16; w1/w3/w2 -> bf16 transposed [N][K] (one-time, bandwidth-bound).
//   Pass 1:   h = silu(x@w1) * (x@w3)  -- WMMA bf16, double-buffered async-LDS staging.
//   Pass 2:   out = h @ w2             -- WMMA bf16, double-buffered async-LDS staging.
// Staging uses global_load_async_to_lds_b128 (ASYNCcnt) pipelined one K-chunk ahead,
// so HBM latency overlaps the WMMA pipe instead of stalling at a barrier.

typedef __bf16 bf16;
typedef __attribute__((ext_vector_type(4)))  __bf16 v4bf;
typedef __attribute__((ext_vector_type(8)))  __bf16 v8bf;
typedef __attribute__((ext_vector_type(16))) __bf16 v16bf;
typedef __attribute__((ext_vector_type(8)))  float  v8f;

#define DDIM 4096
#define IDIM 1024
#define BM 128
#define BN 128
#define BK 64
#define BKP 72   // padded K stride (elements); 144B = 16B multiple

__device__ __forceinline__ v16bf combine16(v8bf lo, v8bf hi) {
    return __builtin_shufflevector(lo, hi, 0, 1, 2, 3, 4, 5, 6, 7,
                                           8, 9, 10, 11, 12, 13, 14, 15);
}

__device__ __forceinline__ unsigned lds_off(const void* p) {
    // Generic LDS address: low 32 bits are the LDS byte offset (ISA 10.2 aperture rules).
    return (unsigned)(unsigned long long)p;
}

// Async 16B copy global -> LDS (per-lane). Tracked with ASYNCcnt.
__device__ __forceinline__ void async_copy_b128(unsigned lds_addr, const void* gptr) {
    asm volatile("global_load_async_to_lds_b128 %0, %1, off"
                 :: "v"(lds_addr), "v"(gptr) : "memory");
}

__device__ __forceinline__ void wait_async0() {
    asm volatile("s_wait_asynccnt 0x0" ::: "memory");
}

// Stage one 128x64 bf16 tile into LDS: 1024 16B chunks, 4 per thread (256 threads).
__device__ __forceinline__ void stage_tile(bf16 (*dst)[BKP], const bf16* src,
                                           size_t src_stride, int t) {
#pragma unroll
    for (int it = 0; it < 4; ++it) {
        const int q = it * 256 + t;
        const int r = q >> 3;           // 0..127
        const int c = (q & 7) * 8;      // 0..56
        async_copy_b128(lds_off(&dst[r][c]), src + (size_t)r * src_stride + c);
    }
}

// Uniform scalar scan: which expert owns token-row `row0`.
__device__ __forceinline__ int find_expert(const int* __restrict__ g, int E, int row0) {
    int off = 0;
    for (int e = 0; e < E; ++e) {
        int c = g[e];
        if (row0 < off + c) return e;
        off += c;
    }
    return E - 1;
}

// ---------------- Pre-pass: fp32 -> bf16 elementwise (x) -----------------
__global__ __launch_bounds__(256) void cvt_bf16_kernel(
    const float* __restrict__ src, bf16* __restrict__ dst)
{
    const size_t i = (size_t)blockIdx.x * blockDim.x + threadIdx.x;
    float4 v = ((const float4*)src)[i];
    v4bf b = { (bf16)v.x, (bf16)v.y, (bf16)v.z, (bf16)v.w };
    ((v4bf*)dst)[i] = b;
}

// ---------------- Pre-pass: [E][R][C] fp32 -> [E][C][R] bf16 -----------------
__global__ __launch_bounds__(256) void transpose_cvt_kernel(
    const float* __restrict__ src, bf16* __restrict__ dst, int R, int C)
{
    __shared__ bf16 tile[32][33];
    const float* s = src + (size_t)blockIdx.z * R * C;
    bf16*        d = dst + (size_t)blockIdx.z * R * C;
    const int r0 = blockIdx.x * 32;
    const int c0 = blockIdx.y * 32;
    const int tx = threadIdx.x & 31;
    const int ty = threadIdx.x >> 5;      // 0..7
#pragma unroll
    for (int i = 0; i < 32; i += 8)
        tile[ty + i][tx] = (bf16)s[(size_t)(r0 + ty + i) * C + c0 + tx];
    __syncthreads();
#pragma unroll
    for (int i = 0; i < 32; i += 8)
        d[(size_t)(c0 + ty + i) * R + r0 + tx] = tile[tx][ty + i];
}

// ---------------- Pass 1: h = silu(x@w1) * (x@w3), bf16 out -----------------
__global__ __launch_bounds__(256) void moe_up_kernel(
    const bf16* __restrict__ xb,    // [T, D] bf16
    const bf16* __restrict__ w1t,   // [E, I, D] bf16 ([N][K])
    const bf16* __restrict__ w3t,   // [E, I, D] bf16 ([N][K])
    const int*  __restrict__ g,     // [E]
    bf16*       __restrict__ h,     // [T, I] bf16 (workspace)
    int E)
{
    __shared__ __align__(16) bf16 As [2][BM][BKP];  // x tile [M][K]
    __shared__ __align__(16) bf16 Bs1[2][BN][BKP];  // w1 tile [N][K]
    __shared__ __align__(16) bf16 Bs3[2][BN][BKP];  // w3 tile [N][K]

    const int t     = threadIdx.x;
    const int lane  = t & 31;
    const int wave  = t >> 5;
    const int wm    = wave & 1;   // 0..1 : 64-row wave tile
    const int wn    = wave >> 1;  // 0..3 : 32-col wave tile
    const int lrow  = lane & 15;
    const int lhalf = lane >> 4;

    const int row0 = blockIdx.x * BM;
    const int col0 = blockIdx.y * BN;

    const int e = find_expert(g, E, row0);
    const bf16* xrow = xb  + (size_t)row0 * DDIM;
    const bf16* w1p  = w1t + (size_t)e * DDIM * IDIM + (size_t)col0 * DDIM;
    const bf16* w3p  = w3t + (size_t)e * DDIM * IDIM + (size_t)col0 * DDIM;

    v8f acc1[4][2], acc3[4][2];
    v8f zero = {};
#pragma unroll
    for (int i = 0; i < 4; ++i)
#pragma unroll
        for (int j = 0; j < 2; ++j) { acc1[i][j] = zero; acc3[i][j] = zero; }

    // Prologue: stage chunk 0 into buffer 0.
    stage_tile(As[0],  xrow, DDIM, t);
    stage_tile(Bs1[0], w1p,  DDIM, t);
    stage_tile(Bs3[0], w3p,  DDIM, t);

    const int nk = DDIM / BK;
    for (int ki = 0; ki < nk; ++ki) {
        wait_async0();       // my copies into buf[ki&1] are done
        __syncthreads();     // everyone's copies done; everyone past compute ki-1
        const int cur = ki & 1;

        // Pipeline: kick off chunk ki+1 into the other buffer while computing.
        if (ki + 1 < nk) {
            const int kb = (ki + 1) * BK;
            stage_tile(As[cur ^ 1],  xrow + kb, DDIM, t);
            stage_tile(Bs1[cur ^ 1], w1p + kb,  DDIM, t);
            stage_tile(Bs3[cur ^ 1], w3p + kb,  DDIM, t);
        }

        // ---- 2 k-steps of 32, WMMA 16x16x32 bf16 ----
#pragma unroll
        for (int s = 0; s < 2; ++s) {
            const int k32 = s * 32;
            v16bf a[4], b1[2], b3[2];
#pragma unroll
            for (int i = 0; i < 4; ++i) {
                const bf16* p = &As[cur][wm * 64 + i * 16 + lrow][k32 + lhalf * 8];
                a[i] = combine16(*(const v8bf*)p, *(const v8bf*)(p + 16));
            }
#pragma unroll
            for (int j = 0; j < 2; ++j) {
                const bf16* p1 = &Bs1[cur][wn * 32 + j * 16 + lrow][k32 + lhalf * 8];
                b1[j] = combine16(*(const v8bf*)p1, *(const v8bf*)(p1 + 16));
                const bf16* p3 = &Bs3[cur][wn * 32 + j * 16 + lrow][k32 + lhalf * 8];
                b3[j] = combine16(*(const v8bf*)p3, *(const v8bf*)(p3 + 16));
            }
#pragma unroll
            for (int i = 0; i < 4; ++i)
#pragma unroll
                for (int j = 0; j < 2; ++j) {
                    acc1[i][j] = __builtin_amdgcn_wmma_f32_16x16x32_bf16(
                        false, a[i], false, b1[j], (short)0, acc1[i][j], false, false);
                    acc3[i][j] = __builtin_amdgcn_wmma_f32_16x16x32_bf16(
                        false, a[i], false, b3[j], (short)0, acc3[i][j], false, false);
                }
        }
    }

    // ---- SwiGLU epilogue, store bf16 h ----
#pragma unroll
    for (int i = 0; i < 4; ++i)
#pragma unroll
        for (int j = 0; j < 2; ++j)
#pragma unroll
            for (int r = 0; r < 8; ++r) {
                const float v1 = acc1[i][j][r];
                const float v3 = acc3[i][j][r];
                const float sv = v1 * __builtin_amdgcn_rcpf(1.0f + __expf(-v1));
                const int m = wm * 64 + i * 16 + r + lhalf * 8;
                const int n = wn * 32 + j * 16 + lrow;
                h[(size_t)(row0 + m) * IDIM + col0 + n] = (bf16)(sv * v3);
            }
}

// ---------------- Pass 2: out = h @ w2, fp32 out -----------------
__global__ __launch_bounds__(256) void moe_down_kernel(
    const bf16* __restrict__ h,     // [T, I] bf16
    const bf16* __restrict__ w2t,   // [E, D, I] bf16 ([N][K])
    const int*  __restrict__ g,     // [E]
    float*      __restrict__ out,   // [T, D] fp32
    int E)
{
    __shared__ __align__(16) bf16 As[2][BM][BKP];  // h tile [M][K]
    __shared__ __align__(16) bf16 Bs[2][BN][BKP];  // w2 tile [N][K]

    const int t     = threadIdx.x;
    const int lane  = t & 31;
    const int wave  = t >> 5;
    const int wm    = wave & 1;
    const int wn    = wave >> 1;
    const int lrow  = lane & 15;
    const int lhalf = lane >> 4;

    const int row0 = blockIdx.x * BM;
    const int col0 = blockIdx.y * BN;

    const int e = find_expert(g, E, row0);
    const bf16* hrow = h   + (size_t)row0 * IDIM;
    const bf16* w2p  = w2t + (size_t)e * IDIM * DDIM + (size_t)col0 * IDIM;

    v8f acc[4][2];
    v8f zero = {};
#pragma unroll
    for (int i = 0; i < 4; ++i)
#pragma unroll
        for (int j = 0; j < 2; ++j) acc[i][j] = zero;

    stage_tile(As[0], hrow, IDIM, t);
    stage_tile(Bs[0], w2p,  IDIM, t);

    const int nk = IDIM / BK;
    for (int ki = 0; ki < nk; ++ki) {
        wait_async0();
        __syncthreads();
        const int cur = ki & 1;

        if (ki + 1 < nk) {
            const int kb = (ki + 1) * BK;
            stage_tile(As[cur ^ 1], hrow + kb, IDIM, t);
            stage_tile(Bs[cur ^ 1], w2p + kb,  IDIM, t);
        }

#pragma unroll
        for (int s = 0; s < 2; ++s) {
            const int k32 = s * 32;
            v16bf a[4], b[2];
#pragma unroll
            for (int i = 0; i < 4; ++i) {
                const bf16* p = &As[cur][wm * 64 + i * 16 + lrow][k32 + lhalf * 8];
                a[i] = combine16(*(const v8bf*)p, *(const v8bf*)(p + 16));
            }
#pragma unroll
            for (int j = 0; j < 2; ++j) {
                const bf16* p = &Bs[cur][wn * 32 + j * 16 + lrow][k32 + lhalf * 8];
                b[j] = combine16(*(const v8bf*)p, *(const v8bf*)(p + 16));
            }
#pragma unroll
            for (int i = 0; i < 4; ++i)
#pragma unroll
                for (int j = 0; j < 2; ++j)
                    acc[i][j] = __builtin_amdgcn_wmma_f32_16x16x32_bf16(
                        false, a[i], false, b[j], (short)0, acc[i][j], false, false);
        }
    }

#pragma unroll
    for (int i = 0; i < 4; ++i)
#pragma unroll
        for (int j = 0; j < 2; ++j)
#pragma unroll
            for (int r = 0; r < 8; ++r) {
                const int m = wm * 64 + i * 16 + r + lhalf * 8;
                const int n = wn * 32 + j * 16 + lrow;
                out[(size_t)(row0 + m) * DDIM + col0 + n] = acc[i][j][r];
            }
}

extern "C" void kernel_launch(void* const* d_in, const int* in_sizes, int n_in,
                              void* d_out, int out_size, void* d_ws, size_t ws_size,
                              hipStream_t stream) {
    const float* x  = (const float*)d_in[0];
    const float* w1 = (const float*)d_in[1];
    const float* w2 = (const float*)d_in[2];
    const float* w3 = (const float*)d_in[3];
    const int*   g  = (const int*)d_in[4];

    const int T = in_sizes[0] / DDIM;   // 32768
    const int E = in_sizes[4];          // 8

    // Workspace layout (bf16 elements): h | xb | w1t | w3t | w2t  (~512 MB total)
    bf16* ws  = (bf16*)d_ws;
    bf16* h   = ws;                         size_t off = (size_t)T * IDIM;
    bf16* xb  = ws + off;                   off += (size_t)T * DDIM;
    bf16* w1t = ws + off;                   off += (size_t)E * DDIM * IDIM;
    bf16* w3t = ws + off;                   off += (size_t)E * DDIM * IDIM;
    bf16* w2t = ws + off;

    float* out = (float*)d_out;
    dim3 block(256, 1, 1);

    // Pre-pass: x -> bf16
    {
        const size_t n4 = (size_t)T * DDIM / 4;
        cvt_bf16_kernel<<<dim3((unsigned)(n4 / 256)), block, 0, stream>>>(x, xb);
    }
    // Pre-pass: weights -> bf16, transposed to [N][K]
    transpose_cvt_kernel<<<dim3(DDIM / 32, IDIM / 32, E), block, 0, stream>>>(w1, w1t, DDIM, IDIM);
    transpose_cvt_kernel<<<dim3(DDIM / 32, IDIM / 32, E), block, 0, stream>>>(w3, w3t, DDIM, IDIM);
    transpose_cvt_kernel<<<dim3(IDIM / 32, DDIM / 32, E), block, 0, stream>>>(w2, w2t, IDIM, DDIM);

    // Main passes
    moe_up_kernel<<<dim3(T / BM, IDIM / BN), block, 0, stream>>>(xb, w1t, w3t, g, h, E);
    moe_down_kernel<<<dim3(T / BM, DDIM / BN), block, 0, stream>>>(h, w2t, g, out, E);
}
